// MultiMLPLayer_62947040690424
// MI455X (gfx1250) — matrix operand order repaired and checked by reading, not compiled, not run on hardware
//
#include <hip/hip_runtime.h>
#include <hip/hip_bf16.h>
#include <math.h>

typedef __attribute__((ext_vector_type(16))) __bf16 v16bf;
typedef __attribute__((ext_vector_type(8)))  float  v8f;
typedef __attribute__((ext_vector_type(4)))  unsigned int v4u;
typedef __attribute__((ext_vector_type(8)))  int v8i;
typedef __attribute__((ext_vector_type(4)))  int v4i;

#define BM 128
#define BN 64
#define BK 32
#define LDA 40   // bf16 elems per A-tile row in LDS (32 + 8 pad = 80B pitch)
#define LDB 40   // bf16 elems per Bt-tile row in LDS (32 + 8 pad)

#if defined(__has_builtin)
#if __has_builtin(__builtin_amdgcn_tensor_load_to_lds) && __has_builtin(__builtin_amdgcn_s_wait_tensorcnt)
#define USE_TDM 1
#endif
#endif
#ifndef USE_TDM
#define USE_TDM 0
#endif

__device__ __forceinline__ float gelu_erf(float v) {
    return 0.5f * v * (1.0f + erff(v * 0.70710678118654752440f));
}

#if USE_TDM
// Issue one TDM 2D load: tile_d0 x tile_d1 bf16 elements, row stride stride0 (elems),
// into LDS at lds_byte_addr with 4-DWORD pad every 16 DWORDs (row pitch 80B = LDA*2).
__device__ __forceinline__ void tdm_load_tile_bf16(unsigned lds_byte_addr, const void* gptr,
                                                   int tile_d0, int tile_d1, long stride0) {
    unsigned long long ga = (unsigned long long)(size_t)gptr;
    v4u g0;
    g0[0] = 1u;                                        // count=1, user mode
    g0[1] = lds_byte_addr;                             // lds_addr
    g0[2] = (unsigned)(ga & 0xFFFFFFFFull);            // global_addr[31:0]
    g0[3] = (unsigned)((ga >> 32) & 0x1FFFFFFull)      // global_addr[56:32]
          | (2u << 30);                                // type = 2 ("image")
    v8i g1;
    g1[0] = (1 << 16)                                  // data_size = 2 bytes
          | (1 << 20)                                  // pad_enable
          | (3 << 22)                                  // pad_interval: every 16 DWORDs (64B row)
          | (3 << 25);                                 // pad_amount: 4 DWORDs (16B)
    g1[1] = (tile_d0 & 0xFFFF) << 16;                  // tensor_dim0 = tile_d0 (tile in-bounds)
    g1[2] = ((unsigned)tile_d0 >> 16) | ((tile_d1 & 0xFFFF) << 16);  // tensor_dim1 = tile_d1
    g1[3] = ((unsigned)tile_d1 >> 16) | (tile_d0 << 16);             // tile_dim0
    g1[4] = tile_d1;                                   // tile_dim1 (tile_dim2 = 0)
    g1[5] = (int)(stride0 & 0xFFFFFFFFll);             // tensor_dim0_stride[31:0]
    g1[6] = (int)((stride0 >> 32) & 0xFFFFll);         // tensor_dim0_stride[47:32]
    g1[7] = 0;
    v4i z4 = {0, 0, 0, 0};
    v8i z8 = {0, 0, 0, 0, 0, 0, 0, 0};
    // amdgpu-toolchain (clang-23) 6-arg form: (g0, g1, g2, g3, extra, cpol)
    __builtin_amdgcn_tensor_load_to_lds(g0, g1, z4, z4, z8, 0);
}
#endif

// ---------------- f32 -> bf16 bulk convert ----------------
__global__ __launch_bounds__(256)
void cvt_f32_bf16(const float* __restrict__ src, __bf16* __restrict__ dst, int n) {
    int i = (blockIdx.x * 256 + threadIdx.x) * 8;
    if (i + 8 > n) return;
    const float4* s = (const float4*)(src + i);
    float4 a = s[0], b = s[1];
    union { __bf16 h[8]; uint4 u; } o;
    o.h[0] = (__bf16)a.x; o.h[1] = (__bf16)a.y; o.h[2] = (__bf16)a.z; o.h[3] = (__bf16)a.w;
    o.h[4] = (__bf16)b.x; o.h[5] = (__bf16)b.y; o.h[6] = (__bf16)b.z; o.h[7] = (__bf16)b.w;
    *(uint4*)(dst + i) = o.u;
}

// ---------------- bf16 WMMA GEMM ----------------
// C[M,N] = A[M,K](bf16) x B[K,N](f32 -> bf16 in LDS, transposed) + bias
// EPI 0: out bf16 = gelu(acc+bias), stride N
// EPI 1: if (eid[m]==EXP) out f32[m*N+n] = alpha * (acc+bias)
template<int EPI, int EXP>
__global__ __launch_bounds__(256)
void gemm_bf16(const __bf16* __restrict__ A, const float* __restrict__ B,
               const float* __restrict__ bias, void* __restrict__ outp,
               const int* __restrict__ eid, const float* __restrict__ alpha_p,
               int M, int N, int K) {
    __shared__ __bf16 sA[BM * LDA];
    __shared__ __bf16 sB[BN * LDB];   // stored transposed: sB[n][k]

    const int tid  = threadIdx.x;
    const int m0   = blockIdx.y * BM;
    const int n0   = blockIdx.x * BN;
    const int wave = tid >> 5;
    const int lane = tid & 31;
    const int h    = lane >> 4;       // half-wave
    const int mr   = lane & 15;
    const int m_base = wave * 16;

    // B loader coords: each thread handles one k-pair for 4 columns
    const int kp = tid >> 4;          // 0..15 -> k = 2*kp, 2*kp+1
    const int n4 = (tid & 15) * 4;    // 0..60

#if USE_TDM
    const unsigned lds_a = (unsigned)(size_t)(void*)&sA[0];
#else
    const int arow = tid >> 1, ahalf = tid & 1;
#endif

    v8f acc[4];
#pragma unroll
    for (int f = 0; f < 4; ++f)
#pragma unroll
        for (int r = 0; r < 8; ++r) acc[f][r] = 0.0f;

    for (int k0 = 0; k0 < K; k0 += BK) {
        // --- A tile: global bf16 -> LDS (TDM DMA with native row padding) ---
#if USE_TDM
        if (tid < 32) {
            tdm_load_tile_bf16(lds_a, A + (size_t)m0 * K + k0, BK, BM, (long)K);
        }
#else
        {
            const uint4* g = (const uint4*)(A + (size_t)(m0 + arow) * K + k0 + ahalf * 16);
            uint4 u0 = g[0], u1 = g[1];
            uint4* su = (uint4*)&sA[arow * LDA + ahalf * 16];
            su[0] = u0; su[1] = u1;
        }
#endif
        // --- B tile: global f32 -> bf16 -> LDS transposed, packed dword stores ---
        {
            const float* gp = B + (size_t)(k0 + 2 * kp) * N + n0 + n4;
            float4 r0 = *(const float4*)gp;
            float4 r1 = *(const float4*)(gp + N);
            float f0[4] = {r0.x, r0.y, r0.z, r0.w};
            float f1[4] = {r1.x, r1.y, r1.z, r1.w};
#pragma unroll
            for (int i = 0; i < 4; ++i) {
                union { __bf16 h[2]; unsigned u; } p;
                p.h[0] = (__bf16)f0[i];
                p.h[1] = (__bf16)f1[i];
                *(unsigned*)&sB[(n4 + i) * LDB + 2 * kp] = p.u;
            }
        }
#if USE_TDM
        if (tid < 32) __builtin_amdgcn_s_wait_tensorcnt(0);
#endif
        __syncthreads();

        // --- A fragment: lane holds row m, K chunks [h*8,h*8+8) and [16+h*8,...) ---
        union { uint4 u[2]; v16bf v; } av;
        const __bf16* ar = &sA[(m_base + mr) * LDA];
        av.u[0] = *(const uint4*)(ar + h * 8);
        av.u[1] = *(const uint4*)(ar + 16 + h * 8);

        // --- B fragments: lane holds col n, K chunks [h*8,h*8+8) and [16+h*8,...) ---
        union { uint4 u[2]; v16bf v; } bv[4];
#pragma unroll
        for (int f = 0; f < 4; ++f) {
            const __bf16* br = &sB[(f * 16 + mr) * LDB + h * 8];
            bv[f].u[0] = *(const uint4*)(br);
            bv[f].u[1] = *(const uint4*)(br + 16);
        }
#pragma unroll
        for (int f = 0; f < 4; ++f) {
            acc[f] = __builtin_amdgcn_wmma_f32_16x16x32_bf16(
                false, av.v, false, bv[f].v, (short)0, acc[f], false, false);
        }
        __syncthreads();
    }

    // --- epilogue ---
    const float alpha = (EPI == 1) ? alpha_p[0] : 0.0f;
#pragma unroll
    for (int f = 0; f < 4; ++f) {
        const int n = n0 + f * 16 + mr;
        const float bv = bias[n];
#pragma unroll
        for (int r = 0; r < 8; ++r) {
            const int m = m0 + m_base + h * 8 + r;   // D layout: VGPR r -> M = 8*half + r
            const float v = acc[f][r] + bv;
            if (EPI == 0) {
                ((__bf16*)outp)[(size_t)m * N + n] = (__bf16)gelu_erf(v);
            } else {
                if (eid[m] == EXP)
                    ((float*)outp)[(size_t)m * N + n] = alpha * v;
            }
        }
    }
}

// ---------------- gate second layer + argmax ----------------
__global__ __launch_bounds__(256)
void gate_logits_argmax(const __bf16* __restrict__ hg, const float* __restrict__ gw2,
                        const float* __restrict__ gb2, const float* __restrict__ ebias,
                        int* __restrict__ eid, int T) {
    __shared__ float sw[256 * 6];
    __shared__ float sb[6];
    for (int i = threadIdx.x; i < 256 * 6; i += 256) sw[i] = gw2[i];
    if (threadIdx.x < 6) sb[threadIdx.x] = gb2[threadIdx.x] + ebias[threadIdx.x];
    __syncthreads();
    const int t = blockIdx.x * 256 + threadIdx.x;
    if (t >= T) return;
    const __bf16* hr = hg + (size_t)t * 256;
    float l[6] = {sb[0], sb[1], sb[2], sb[3], sb[4], sb[5]};
    for (int g = 0; g < 256; ++g) {
        const float hv = (float)hr[g];
#pragma unroll
        for (int m = 0; m < 6; ++m) l[m] += hv * sw[g * 6 + m];
    }
    int best = 0; float bvv = l[0];
#pragma unroll
    for (int m = 1; m < 6; ++m)
        if (l[m] > bvv) { bvv = l[m]; best = m; }   // strict > : first max wins (jax top_k tie rule)
    eid[t] = best;
}

// ---------------- cheap experts 2..5 (rank-2/rank-4/ReGLU/FiLM) ----------------
__global__ __launch_bounds__(256)
void cheap_experts(const float* __restrict__ x, const int* __restrict__ eid,
                   float* __restrict__ out, const float* __restrict__ pm,
                   const float* __restrict__ p2_w, const float* __restrict__ p2_v,
                   const float* __restrict__ p2_alpha, const float* __restrict__ p2_b,
                   const float* __restrict__ p2_bias,
                   const float* __restrict__ p4_w, const float* __restrict__ p4_v,
                   const float* __restrict__ p4_alpha, const float* __restrict__ p4_b,
                   const float* __restrict__ p4_bias,
                   const float* __restrict__ rg_u, const float* __restrict__ rg_a,
                   const float* __restrict__ rg_b, const float* __restrict__ rg_bias,
                   const float* __restrict__ fl_dw, const float* __restrict__ fl_db,
                   const float* __restrict__ fl_uw, const float* __restrict__ fl_ub) {
    const int t = blockIdx.x;
    const int e = eid[t];
    if (e < 2) return;                       // experts 0/1 handled by GEMM epilogues
    const int tid = threadIdx.x;
    const float* xt = x + (size_t)t * 1024;

    float xv[4];
#pragma unroll
    for (int j = 0; j < 4; ++j) xv[j] = xt[tid + 256 * j];

    float ps[16];
#pragma unroll
    for (int c = 0; c < 16; ++c) ps[c] = 0.0f;
    const int nch = (e == 2) ? 2 : (e == 3) ? 4 : (e == 4) ? 1 : 16;

    if (e == 2) {
#pragma unroll
        for (int j = 0; j < 4; ++j) {
            const int i = tid + 256 * j;
            ps[0] += xv[j] * p2_w[i];
            ps[1] += xv[j] * p2_w[1024 + i];
        }
    } else if (e == 3) {
#pragma unroll
        for (int j = 0; j < 4; ++j) {
            const int i = tid + 256 * j;
#pragma unroll
            for (int r = 0; r < 4; ++r) ps[r] += xv[j] * p4_w[r * 1024 + i];
        }
    } else if (e == 4) {
#pragma unroll
        for (int j = 0; j < 4; ++j) ps[0] += xv[j] * rg_u[tid + 256 * j];
    } else {
#pragma unroll
        for (int j = 0; j < 4; ++j) {
            const int i = tid + 256 * j;
            const float* dr = fl_dw + (size_t)i * 16;
#pragma unroll
            for (int c = 0; c < 16; ++c) ps[c] += xv[j] * dr[c];
        }
    }

    __shared__ float red[16 * 256];
    for (int c = 0; c < nch; ++c) red[c * 256 + tid] = ps[c];
    __syncthreads();
    for (int s = 128; s > 0; s >>= 1) {
        if (tid < s)
            for (int c = 0; c < nch; ++c) red[c * 256 + tid] += red[c * 256 + tid + s];
        __syncthreads();
    }

    __shared__ float coef[16];
    __shared__ float sc0;
    if (tid == 0) {
        if (e == 2) {
            for (int i = 0; i < 2; ++i) coef[i] = p2_alpha[i] * gelu_erf(red[i * 256] + p2_b[i]);
        } else if (e == 3) {
            for (int i = 0; i < 4; ++i) coef[i] = p4_alpha[i] * gelu_erf(red[i * 256] + p4_b[i]);
        } else if (e == 4) {
            sc0 = 1.0f / (1.0f + expf(-(red[0] + rg_b[0])));
        } else {
            for (int c = 0; c < 16; ++c) coef[c] = gelu_erf(red[c * 256] + fl_db[c]);
        }
    }
    __syncthreads();

    const float pa = pm[0];
#pragma unroll
    for (int j = 0; j < 4; ++j) {
        const int i = tid + 256 * j;
        float o;
        if (e == 2) {
            o = p2_bias[i] + coef[0] * p2_v[i] + coef[1] * p2_v[1024 + i];
        } else if (e == 3) {
            o = p4_bias[i];
#pragma unroll
            for (int r = 0; r < 4; ++r) o += coef[r] * p4_v[r * 1024 + i];
        } else if (e == 4) {
            o = sc0 * xv[j] * rg_a[i] + rg_bias[i];
        } else {
            float gm = fl_ub[i], bt = fl_ub[1024 + i];
#pragma unroll
            for (int c = 0; c < 16; ++c) {
                gm += coef[c] * fl_uw[c * 2048 + i];
                bt += coef[c] * fl_uw[c * 2048 + 1024 + i];
            }
            o = gm * xv[j] + bt;
        }
        out[(size_t)t * 1024 + i] = pa * o;
    }
}

extern "C" void kernel_launch(void* const* d_in, const int* in_sizes, int n_in,
                              void* d_out, int out_size, void* d_ws, size_t ws_size,
                              hipStream_t stream) {
    const float* x       = (const float*)d_in[0];
    const float* gw1     = (const float*)d_in[1];
    const float* gb1     = (const float*)d_in[2];
    const float* gw2     = (const float*)d_in[3];
    const float* gb2     = (const float*)d_in[4];
    const float* ebias   = (const float*)d_in[5];
    const float* pm      = (const float*)d_in[6];
    const float* dfc_w   = (const float*)d_in[7];
    const float* dfc_b   = (const float*)d_in[8];
    const float* dproj_w = (const float*)d_in[9];
    const float* dproj_b = (const float*)d_in[10];
    const float* sfc_w   = (const float*)d_in[11];
    const float* sfc_b   = (const float*)d_in[12];
    const float* sproj_w = (const float*)d_in[13];
    const float* sproj_b = (const float*)d_in[14];
    const float* p2_w    = (const float*)d_in[15];
    const float* p2_v    = (const float*)d_in[16];
    const float* p2_alpha= (const float*)d_in[17];
    const float* p2_b    = (const float*)d_in[18];
    const float* p2_bias = (const float*)d_in[19];
    const float* p4_w    = (const float*)d_in[20];
    const float* p4_v    = (const float*)d_in[21];
    const float* p4_alpha= (const float*)d_in[22];
    const float* p4_b    = (const float*)d_in[23];
    const float* p4_bias = (const float*)d_in[24];
    const float* rg_u    = (const float*)d_in[25];
    const float* rg_a    = (const float*)d_in[26];
    const float* rg_b    = (const float*)d_in[27];
    const float* rg_bias = (const float*)d_in[28];
    const float* fl_dw   = (const float*)d_in[29];
    const float* fl_db   = (const float*)d_in[30];
    const float* fl_uw   = (const float*)d_in[31];
    const float* fl_ub   = (const float*)d_in[32];
    float* out = (float*)d_out;

    const int T = 4 * 2048;   // 8192 tokens
    const int E = 1024;

    char* ws = (char*)d_ws;
    __bf16* xb   = (__bf16*)ws;                                   // 16 MB: x in bf16
    __bf16* hg   = (__bf16*)(ws + (size_t)16 * 1024 * 1024);      //  4 MB: gate hidden
    int*    eid  = (int*)   (ws + (size_t)20 * 1024 * 1024);      // 32 KB: routed expert
    __bf16* hbuf = (__bf16*)(ws + (size_t)21 * 1024 * 1024);      // 32 MB: expert hidden

    // 1) x -> bf16
    cvt_f32_bf16<<<(T * E / 8 + 255) / 256, 256, 0, stream>>>(x, xb, T * E);

    // 2) gate layer 1: hg = gelu(x @ gw1 + gb1)  [8192 x 256]
    gemm_bf16<0, 0><<<dim3(256 / BN, T / BM), 256, 0, stream>>>(
        xb, gw1, gb1, (void*)hg, nullptr, nullptr, T, 256, 1024);

    // 3) logits + argmax -> eid
    gate_logits_argmax<<<T / 256, 256, 0, stream>>>(hg, gw2, gb2, ebias, eid, T);

    // 4) expert0 fc: h0 = gelu(x @ dfc_w + dfc_b)  [8192 x 2048]
    gemm_bf16<0, 0><<<dim3(2048 / BN, T / BM), 256, 0, stream>>>(
        xb, dfc_w, dfc_b, (void*)hbuf, nullptr, nullptr, T, 2048, 1024);

    // 5) expert0 proj, masked store (eid==0), scaled by pm_alpha
    gemm_bf16<1, 0><<<dim3(1024 / BN, T / BM), 256, 0, stream>>>(
        hbuf, dproj_w, dproj_b, (void*)out, eid, pm, T, 1024, 2048);

    // 6) expert1 fc: h1 = gelu(x @ sfc_w + sfc_b)  [8192 x 1024] (reuse hbuf)
    gemm_bf16<0, 0><<<dim3(1024 / BN, T / BM), 256, 0, stream>>>(
        xb, sfc_w, sfc_b, (void*)hbuf, nullptr, nullptr, T, 1024, 1024);

    // 7) expert1 proj, masked store (eid==1)
    gemm_bf16<1, 1><<<dim3(1024 / BN, T / BM), 256, 0, stream>>>(
        hbuf, sproj_w, sproj_b, (void*)out, eid, pm, T, 1024, 1024);

    // 8) cheap experts (eid in {2,3,4,5})
    cheap_experts<<<T, 256, 0, stream>>>(
        x, eid, out, pm,
        p2_w, p2_v, p2_alpha, p2_b, p2_bias,
        p4_w, p4_v, p4_alpha, p4_b, p4_bias,
        rg_u, rg_a, rg_b, rg_bias,
        fl_dw, fl_db, fl_uw, fl_ub);
}
